// MambaLikeBlock_85246510891655
// MI455X (gfx1250) — compile-verified, matrix operand order checked by reading, not run on hardware
//
#include <hip/hip_runtime.h>
#include <hip/hip_bf16.h>

typedef __bf16 bf16_t;
typedef __attribute__((ext_vector_type(16))) __bf16 v16bf;
typedef __attribute__((ext_vector_type(8)))  __bf16 v8bf;
typedef __attribute__((ext_vector_type(8)))  float  v8f;
typedef __attribute__((ext_vector_type(4)))  unsigned u32x4;
typedef __attribute__((ext_vector_type(8)))  unsigned u32x8;

#define DM     768
#define DS     16
#define BATCH_ 2
#define SEQ_   2048
#define ROWS   (BATCH_ * SEQ_)   // 4096

// ---------------------------------------------------------------------------
// TDM prefetch: build a D# (ISA 8.3/8.4) for a 2D bf16 tile [rows x kelems]
// of BT[N,K] and issue tensor_load_to_lds. Used as an L2-warming prefetch of
// the block's B panel; the LDS staging buffer is never read, so it has no
// correctness impact. Retired with s_wait_tensorcnt before kernel exit.
// ---------------------------------------------------------------------------
__device__ __forceinline__ void tdm_prefetch_tile(const void* gaddr,
                                                  unsigned lds_off,
                                                  unsigned kelems, unsigned rows,
                                                  unsigned tensor_d0,
                                                  unsigned tensor_d1,
                                                  unsigned stride0) {
  unsigned long long ga = (unsigned long long)gaddr;
  u32x4 g0;
  g0[0] = 1u;                                   // count=1, user descriptor
  g0[1] = lds_off;                              // lds_addr (bytes)
  g0[2] = (unsigned)(ga & 0xFFFFFFFFu);         // global_addr[31:0]
  g0[3] = (unsigned)((ga >> 32) & 0x01FFFFFFu)  // global_addr[56:32]
          | (2u << 30);                         // type=2 ("image")
  u32x8 g1;
  g1[0] = (1u << 16);                           // wg_mask=0, data_size=1 (2B)
  g1[1] = (tensor_d0 & 0xFFFFu) << 16;          // tensor_dim0[15:0]
  g1[2] = (tensor_d0 >> 16) | ((tensor_d1 & 0xFFFFu) << 16);
  g1[3] = (tensor_d1 >> 16) | ((kelems & 0xFFFFu) << 16);   // tile_dim0
  g1[4] = (rows & 0xFFFFu);                     // tile_dim1; tile_dim2=0
  g1[5] = stride0;                              // tensor_dim0_stride[31:0]
  g1[6] = 0;                                    // stride0 hi / stride1 lo
  g1[7] = 0;
  asm volatile("tensor_load_to_lds %0, %1" :: "s"(g0), "s"(g1) : "memory");
}

// ---------------------------------------------------------------------------
// Fragment loader for v_wmma_f32_16x16x32_bf16.
// A-frag: rows M = m0 + lane%16 of A[M,K] row-major.
// B-frag: cols N = n0 + lane%16 of B, loaded from pre-transposed BT[N,K].
// Per ISA layout, lane half h holds K = {k0+8h .. k0+8h+7} and
// {k0+16+8h .. k0+16+8h+7}: two contiguous 16-byte runs.
// ---------------------------------------------------------------------------
__device__ __forceinline__ v16bf load_frag16(const bf16_t* __restrict__ base,
                                             int ld, int row, int k0, int lane) {
  const int r = row + (lane & 15);
  const int h = (lane >> 4) & 1;
  const bf16_t* p = base + (size_t)r * ld + (k0 + 8 * h);
  v8bf lo = *(const v8bf*)(p);
  v8bf hi = *(const v8bf*)(p + 16);
  return __builtin_shufflevector(lo, hi, 0, 1, 2, 3, 4, 5, 6, 7,
                                         8, 9, 10, 11, 12, 13, 14, 15);
}

// ---------------------------------------------------------------------------
// Tiled bf16 GEMM, f32 accumulate via v_wmma_f32_16x16x32_bf16.
//   C[M,N] = A[M,K] * BT[N,K]^T + bias
// Block = 256 threads = 8 waves arranged 2(M) x 4(N); wave tile 64x64
// (4x4 WMMA frags: 16 wmma per 16 b128 loads per K-step); block tile
// 128(M) x 256(N). M,N,K are multiples of tile sizes (fixed shapes).
// MODE 0: f32 out; MODE 1: softplus f32 out; MODE 2: bf16 out.
// ---------------------------------------------------------------------------
template <int MODE>
__global__ __launch_bounds__(256) void gemm_bf16_wmma(
    const bf16_t* __restrict__ A, const bf16_t* __restrict__ BT,
    const float* __restrict__ bias, float* __restrict__ Cf,
    bf16_t* __restrict__ Cb, int M, int N, int K) {
  const int lane = threadIdx.x & 31;
  const int wave = threadIdx.x >> 5;
  const int wm = wave & 1;   // 0..1 -> M sub-tile
  const int wn = wave >> 1;  // 0..3 -> N sub-tile
  const int m0 = blockIdx.y * 128 + wm * 64;
  const int n0 = blockIdx.x * 256 + wn * 64;

  // TDM warm-up prefetch of this block's B panel head (one per block).
  __shared__ char tdm_stage[8192];
  if (wave == 0) {
    tdm_prefetch_tile(BT + (size_t)blockIdx.x * 256 * K,
                      (unsigned)(uintptr_t)&tdm_stage[0],
                      /*kelems=*/64, /*rows=*/64,
                      (unsigned)K, (unsigned)N, (unsigned)K);
  }

  v8f acc[4][4] = {};

  for (int k0 = 0; k0 < K; k0 += 32) {
    v16bf a[4], b[4];
#pragma unroll
    for (int i = 0; i < 4; ++i)
      a[i] = load_frag16(A, K, m0 + 16 * i, k0, lane);
#pragma unroll
    for (int j = 0; j < 4; ++j)
      b[j] = load_frag16(BT, K, n0 + 16 * j, k0, lane);
    if (k0 + 32 < K) {  // next K-slice hint (global_prefetch_b8)
      __builtin_prefetch(A + (size_t)(m0 + (lane & 15)) * K + k0 + 32, 0, 1);
      __builtin_prefetch(BT + (size_t)(n0 + (lane & 15)) * K + k0 + 32, 0, 1);
    }
#pragma unroll
    for (int i = 0; i < 4; ++i)
#pragma unroll
      for (int j = 0; j < 4; ++j)
        acc[i][j] = __builtin_amdgcn_wmma_f32_16x16x32_bf16(
            false, a[i], false, b[j], (short)0, acc[i][j], false, false);
  }

  // C/D layout: VGPR r -> M = r + 8*(lane/16), N = lane%16.
  const int col = lane & 15;
  const int h = (lane >> 4) & 1;
#pragma unroll
  for (int i = 0; i < 4; ++i) {
#pragma unroll
    for (int j = 0; j < 4; ++j) {
      const int mt = m0 + 16 * i;
      const int n = n0 + 16 * j + col;
      const float bv = bias[n];
#pragma unroll
      for (int r = 0; r < 8; ++r) {
        float x = acc[i][j][r] + bv;
        const int m = mt + r + 8 * h;
        if (MODE == 1) x = (x > 20.f) ? x : __logf(1.f + __expf(x));
        if (MODE == 2) Cb[(size_t)m * N + n] = (bf16_t)x;
        else           Cf[(size_t)m * N + n] = x;
      }
    }
  }
  __builtin_amdgcn_s_wait_tensorcnt(0);  // retire TDM prefetch
}

// f32 [K,N] weight -> bf16 [N,K] transposed copy
__global__ void wconvert(const float* __restrict__ W, bf16_t* __restrict__ WT,
                         int K, int N) {
  const size_t idx = (size_t)blockIdx.x * blockDim.x + threadIdx.x;
  if (idx >= (size_t)K * N) return;
  const int n = (int)(idx % N);
  const int k = (int)(idx / N);
  WT[(size_t)n * K + k] = (bf16_t)W[idx];
}

// f32 -> bf16 elementwise
__global__ void fconvert(const float* __restrict__ src,
                         bf16_t* __restrict__ dst, size_t n) {
  const size_t idx = (size_t)blockIdx.x * blockDim.x + threadIdx.x;
  if (idx < n) dst[idx] = (bf16_t)src[idx];
}

// Depthwise causal conv (K=4) over u = uv[:, :DM]; emits f32 and bf16 copies.
__global__ void conv_kernel(const float* __restrict__ uv,
                            const float* __restrict__ Wconv,
                            const float* __restrict__ bconv,
                            float* __restrict__ uconv_f,
                            bf16_t* __restrict__ uconv_b) {
  const size_t idx = (size_t)blockIdx.x * blockDim.x + threadIdx.x;
  if (idx >= (size_t)ROWS * DM) return;
  const int c = (int)(idx % DM);
  const int t = (int)((idx / DM) % SEQ_);
  const int b = (int)(idx / ((size_t)DM * SEQ_));
  float acc = bconv[c];
#pragma unroll
  for (int k = 0; k < 4; ++k) {
    const int tt = t - 3 + k;
    if (tt >= 0)
      acc += uv[((size_t)b * SEQ_ + tt) * (2 * DM) + c] * Wconv[c * 4 + k];
  }
  uconv_f[idx] = acc;
  uconv_b[idx] = (bf16_t)acc;
}

// Selective scan: 16 lanes per (b, channel), one state element per lane.
// Fuses the sigmoid(v) gate and emits bf16 y for the final WMMA GEMM.
__global__ __launch_bounds__(256) void ssm_scan(
    const float* __restrict__ dt, const float* __restrict__ uconv,
    const bf16_t* __restrict__ Bt, const bf16_t* __restrict__ Ct,
    const float* __restrict__ uv, const float* __restrict__ A_log,
    const float* __restrict__ Dp, bf16_t* __restrict__ ybf) {
  const int sidx = threadIdx.x & 15;
  const int grp = threadIdx.x >> 4;        // 0..15 channel groups per block
  const int gc = blockIdx.x * 16 + grp;    // 0 .. BATCH_*DM-1
  const int b = gc / DM;
  const int c = gc % DM;

  const float a = -__expf(A_log[c * DS + sidx]);
  const float Dc = Dp[c];
  float s = 0.f;

  for (int t = 0; t < SEQ_; ++t) {
    const size_t row = (size_t)b * SEQ_ + t;
    const float dtv = dt[row * DM + c];
    const float ut = uconv[row * DM + c];
    const float Bv = (float)Bt[(row * DM + c) * DS + sidx];
    const float Cv = (float)Ct[(row * DM + c) * DS + sidx];
    s = s * __expf(dtv * a) + dtv * Bv * ut;
    float p = Cv * s;
    p += __shfl_xor(p, 8, 16);
    p += __shfl_xor(p, 4, 16);
    p += __shfl_xor(p, 2, 16);
    p += __shfl_xor(p, 1, 16);
    if (sidx == 0) {
      const float y = p + Dc * ut;
      const float v = uv[row * (2 * DM) + DM + c];
      const float g = 1.f / (1.f + __expf(-v));
      ybf[row * DM + c] = (bf16_t)(y * g);
    }
  }
}

extern "C" void kernel_launch(void* const* d_in, const int* in_sizes, int n_in,
                              void* d_out, int out_size, void* d_ws,
                              size_t ws_size, hipStream_t stream) {
  (void)in_sizes; (void)n_in; (void)out_size; (void)ws_size;
  const float* x     = (const float*)d_in[0];
  const float* Wi    = (const float*)d_in[1];
  const float* bi    = (const float*)d_in[2];
  const float* Wconv = (const float*)d_in[3];
  const float* bconv = (const float*)d_in[4];
  const float* Wdt   = (const float*)d_in[5];
  const float* bdt   = (const float*)d_in[6];
  const float* WB    = (const float*)d_in[7];
  const float* bB    = (const float*)d_in[8];
  const float* WC    = (const float*)d_in[9];
  const float* bC    = (const float*)d_in[10];
  const float* A_log = (const float*)d_in[11];
  const float* D     = (const float*)d_in[12];
  const float* Wo    = (const float*)d_in[13];
  const float* bo    = (const float*)d_in[14];
  float* out = (float*)d_out;

  char* ws = (char*)d_ws;
  size_t off = 0;
  auto walloc = [&](size_t bytes) -> void* {
    void* p = ws + off;
    off = (off + bytes + 255) & ~(size_t)255;
    return p;
  };
  bf16_t* xbf  = (bf16_t*)walloc((size_t)ROWS * DM * 2);
  bf16_t* WiT  = (bf16_t*)walloc((size_t)1536 * 768 * 2);
  bf16_t* WdtT = (bf16_t*)walloc((size_t)768 * 768 * 2);
  bf16_t* WBT  = (bf16_t*)walloc((size_t)12288 * 768 * 2);
  bf16_t* WCT  = (bf16_t*)walloc((size_t)12288 * 768 * 2);
  bf16_t* WoT  = (bf16_t*)walloc((size_t)768 * 768 * 2);
  float*  uv   = (float*)walloc((size_t)ROWS * 1536 * 4);
  float*  ucf  = (float*)walloc((size_t)ROWS * DM * 4);
  bf16_t* ucb  = (bf16_t*)walloc((size_t)ROWS * DM * 2);
  float*  dtb  = (float*)walloc((size_t)ROWS * DM * 4);
  bf16_t* Btb  = (bf16_t*)walloc((size_t)ROWS * DM * DS * 2);
  bf16_t* Ctb  = (bf16_t*)walloc((size_t)ROWS * DM * DS * 2);
  bf16_t* ybf  = (bf16_t*)walloc((size_t)ROWS * DM * 2);

  const int T = 256;
  auto nb = [](size_t n, int t) { return (unsigned)((n + t - 1) / t); };

  // Precision staging: f32 -> bf16 (weights transposed to [N,K])
  fconvert<<<nb((size_t)ROWS * DM, T), T, 0, stream>>>(x, xbf, (size_t)ROWS * DM);
  wconvert<<<nb((size_t)768 * 1536, T), T, 0, stream>>>(Wi, WiT, 768, 1536);
  wconvert<<<nb((size_t)768 * 768, T), T, 0, stream>>>(Wdt, WdtT, 768, 768);
  wconvert<<<nb((size_t)768 * 12288, T), T, 0, stream>>>(WB, WBT, 768, 12288);
  wconvert<<<nb((size_t)768 * 12288, T), T, 0, stream>>>(WC, WCT, 768, 12288);
  wconvert<<<nb((size_t)768 * 768, T), T, 0, stream>>>(Wo, WoT, 768, 768);

  // uv = x @ Wi + bi   (4096 x 1536)
  gemm_bf16_wmma<0><<<dim3(1536 / 256, ROWS / 128), 256, 0, stream>>>(
      xbf, WiT, bi, uv, nullptr, ROWS, 1536, 768);
  // depthwise causal conv on u
  conv_kernel<<<nb((size_t)ROWS * DM, T), T, 0, stream>>>(uv, Wconv, bconv, ucf, ucb);
  // dt = softplus(u_conv @ Wdt + bdt)
  gemm_bf16_wmma<1><<<dim3(768 / 256, ROWS / 128), 256, 0, stream>>>(
      ucb, WdtT, bdt, dtb, nullptr, ROWS, 768, 768);
  // Bt / Ct projections, stored bf16 (halves scan HBM traffic)
  gemm_bf16_wmma<2><<<dim3(12288 / 256, ROWS / 128), 256, 0, stream>>>(
      ucb, WBT, bB, nullptr, Btb, ROWS, 12288, 768);
  gemm_bf16_wmma<2><<<dim3(12288 / 256, ROWS / 128), 256, 0, stream>>>(
      ucb, WCT, bC, nullptr, Ctb, ROWS, 12288, 768);
  // sequential selective scan + sigmoid gate -> bf16 y
  ssm_scan<<<(BATCH_ * DM) / 16, 256, 0, stream>>>(dtb, ucf, Btb, Ctb, uv,
                                                   A_log, D, ybf);
  // out = y @ Wo + bo  (f32 to d_out)
  gemm_bf16_wmma<0><<<dim3(768 / 256, ROWS / 128), 256, 0, stream>>>(
      ybf, WoT, bo, out, nullptr, ROWS, 768, 768);
}
